// mix_model_2396591751648
// MI455X (gfx1250) — matrix operand order repaired and checked by reading, not compiled, not run on hardware
//
#include <hip/hip_runtime.h>
#include <hip/hip_bf16.h>
#include <math.h>

typedef __bf16 bf16_t;
typedef bf16_t v16bf __attribute__((ext_vector_type(16)));
typedef bf16_t v8bf  __attribute__((ext_vector_type(8)));
typedef bf16_t v2bf  __attribute__((ext_vector_type(2)));
typedef float  v8f   __attribute__((ext_vector_type(8)));

#define NT 256          // threads per block (8 wave32)
#define T2 128
#define FIN 318
#define KP 320          // padded K for GEMM1

// Packed-weight layout offsets (bf16 elements) inside the workspace:
//   Wa: 30 frags (10 kt x 3 nt) = 15360   @ 0
//   Wb:  4 frags (2 kt x 2 nt)  =  2048   @ 15360
//   Wc:  2 frags (1 kt x 2 nt)  =  1024   @ 17408
//   P1:  2 frags (1 kt x 2 nt)  =  1024   @ 18432
#define WPACK_ELEMS 19456

__device__ inline float leaky(float v) { return v > 0.f ? v : 0.1f * v; }

__device__ inline v8f wmma_bf16(v16bf a, v16bf b, v8f c) {
  return __builtin_amdgcn_wmma_f32_16x16x32_bf16(
      /*neg_a=*/false, a, /*neg_b=*/false, b,
      /*c_mod=*/(short)0, c, /*reuse_a=*/false, /*reuse_b=*/false);
}

// A fragment: 16x32 tile from row-major LDS (leading dim ld, all offsets
// multiples of 8 elements -> 16B aligned). Per-lane data is one contiguous
// 16-element run k0+ko+[0..15] (ko = 16 for upper half-wave), matching B packing.
__device__ inline v16bf load_a_frag(const bf16_t* base, int row0, int ld,
                                    int k0, int lane) {
  int m  = lane & 15;
  int ko = (lane < 16) ? 0 : 16;
  const bf16_t* r = base + (row0 + m) * ld + k0 + ko;
  v8bf lo = *(const v8bf*)(r);
  v8bf hi = *(const v8bf*)(r + 8);
  return __builtin_shufflevector(lo, hi, 0, 1, 2, 3, 4, 5, 6, 7, 8, 9, 10, 11,
                                 12, 13, 14, 15);
}

// B fragment from fragment-layout global memory (L2-resident): each lane's
// 16 bf16 are contiguous -> two coalesced global_load_b128 per lane.
__device__ inline v16bf load_b_packed(const bf16_t* __restrict__ packed,
                                      int frag, int lane) {
  const bf16_t* p = packed + ((frag * 32) + lane) * 16;
  v8bf lo = *(const v8bf*)(p);
  v8bf hi = *(const v8bf*)(p + 8);
  return __builtin_shufflevector(lo, hi, 0, 1, 2, 3, 4, 5, 6, 7, 8, 9, 10, 11,
                                 12, 13, 14, 15);
}

// Destination index within a packed fragment array for element (k, n) of a
// KxN weight tile grid with ntiles N-tiles: frag = (k>>5)*ntiles + (n>>4),
// lane = (n&15) + 16*((k>>4)&1), elem = k&15.
__device__ inline int pack_idx(int k, int n, int ntiles) {
  int frag = (k >> 5) * ntiles + (n >> 4);
  int lane = (n & 15) + ((k & 16) ? 16 : 0);
  int e    = k & 15;
  return (frag * 32 + lane) * 16 + e;
}

// ---------------- Setup: pack all weights once (runs as 1 block) ----------
__global__ __launch_bounds__(NT) void pack_weights_kernel(
    const float* __restrict__ Wa, const float* __restrict__ Wb,
    const float* __restrict__ Wc, const float* __restrict__ p1W1,
    bf16_t* __restrict__ wpack) {
  const int tid = threadIdx.x;
  bf16_t* pWa = wpack;
  bf16_t* pWb = wpack + 15360;
  bf16_t* pWc = wpack + 17408;
  bf16_t* pP1 = wpack + 18432;

  for (int i = tid; i < 320 * 48; i += NT) {
    int k = i / 48, n = i - k * 48;
    float v = (k < FIN && n < 40) ? Wa[k * 40 + n] : 0.f;
    pWa[pack_idx(k, n, 3)] = (bf16_t)v;
  }
  for (int i = tid; i < 64 * 32; i += NT) {
    int k = i >> 5, n = i & 31;
    float v = (k < 40 && n < 30) ? Wb[k * 30 + n] : 0.f;
    pWb[pack_idx(k, n, 2)] = (bf16_t)v;
  }
  for (int i = tid; i < 32 * 32; i += NT) {
    int k = i >> 5, n = i & 31;
    float v = (k < 30 && n < 20) ? Wc[k * 20 + n] : 0.f;
    pWc[pack_idx(k, n, 2)] = (bf16_t)v;
  }
  for (int i = tid; i < 32 * 32; i += NT) {
    int k = i >> 5, n = i & 31;
    float v = (k < 20 && n < 30) ? p1W1[k * 30 + n] : 0.f;
    pP1[pack_idx(k, n, 2)] = (bf16_t)v;
  }
}

// ---------------- Stage A: per-(b,t1) tile -> pooled 20-vector ----------------
// LDS (smem, 81920 B), region reused after GEMM1 reads complete:
//   phase 1: sA bf16 [128][320]                                  (81920 B)
//   phase 2: h1 bf16 [128][64] @0, h2 bf16 [128][32] @16384,
//            h3b bf16 [128][32] @24576, h3f f32 [128][20] @32768,
//            s1 f32 [128][32] @43008, scores f32 [128] @59392, red @59904
__global__ __launch_bounds__(NT) void stage_a_kernel(
    const float* __restrict__ x, const bf16_t* __restrict__ wpack,
    const float* __restrict__ ba, const float* __restrict__ bb,
    const float* __restrict__ bc, const float* __restrict__ p1b1,
    const float* __restrict__ p1W2, const float* __restrict__ p1b2,
    float* __restrict__ hpool) {
  const int tid  = threadIdx.x;
  const int lane = tid & 31;
  const int wave = tid >> 5;
  const int blk  = blockIdx.x;          // b*64 + t1

  __shared__ __align__(16) char smem[81920];
  bf16_t* sA = (bf16_t*)smem;           // 128 x 320

  const bf16_t* pWa = wpack;
  const bf16_t* pWb = wpack + 15360;
  const bf16_t* pWc = wpack + 17408;
  const bf16_t* pP1 = wpack + 18432;

  // Stage the 128x318 fp32 x-tile -> bf16 LDS, float2 -> packed bf16x2.
  {
    const float2* xb2 =
        (const float2*)(x + (size_t)blk * (T2 * FIN));  // 128*159 float2
    for (int i = tid; i < T2 * (FIN / 2); i += NT) {
      int r = i / (FIN / 2);
      int c = i - r * (FIN / 2);
      float2 v = xb2[i];
      v2bf o = {(bf16_t)v.x, (bf16_t)v.y};
      *(v2bf*)(&sA[r * KP + 2 * c]) = o;
    }
    if (tid < T2) {  // zero K-pad cols 318..319
      v2bf z = {(bf16_t)0.f, (bf16_t)0.f};
      *(v2bf*)(&sA[tid * KP + FIN]) = z;
    }
  }
  __syncthreads();

  const int row0  = wave * 16;          // each wave owns one 16-row M tile
  const int m_add = (lane < 16) ? 0 : 8;
  const int nlo   = lane & 15;

  // ---- GEMM1: [128x320] @ [320x48] -> 128x40 (acc in VGPRs) ----
  v8f a0 = {}, a1 = {}, a2 = {};
  for (int ks = 0; ks < 10; ++ks) {
    v16bf af = load_a_frag(sA, row0, KP, ks * 32, lane);
    a0 = wmma_bf16(af, load_b_packed(pWa, ks * 3 + 0, lane), a0);
    a1 = wmma_bf16(af, load_b_packed(pWa, ks * 3 + 1, lane), a1);
    a2 = wmma_bf16(af, load_b_packed(pWa, ks * 3 + 2, lane), a2);
  }
  __syncthreads();   // all sA reads done; region is now reusable

  // Zero h1/h2/h3b/h3f regions (padding columns must be 0).
  {
    float* zf = (float*)smem;
    for (int i = tid; i < 10752; i += NT) zf[i] = 0.f;   // 43008 bytes
  }
  __syncthreads();

  bf16_t* h1  = (bf16_t*)smem;            // 128 x 64
  bf16_t* h2  = (bf16_t*)(smem + 16384);  // 128 x 32
  bf16_t* h3b = (bf16_t*)(smem + 24576);  // 128 x 32
  float*  h3f = (float*)(smem + 32768);   // 128 x 20
  float*  s1  = (float*)(smem + 43008);   // 128 x 32
  float*  sc  = (float*)(smem + 59392);   // 128
  float*  red = (float*)(smem + 59904);   // scratch

  // Store h1 = leaky(acc + ba), bf16, row-major ld 64.
  {
    v8f accs[3] = {a0, a1, a2};
#pragma unroll
    for (int t = 0; t < 3; ++t) {
      int nn = t * 16 + nlo;
      if (nn < 40) {
        float bv = ba[nn];
#pragma unroll
        for (int v = 0; v < 8; ++v) {
          float val = leaky(accs[t][v] + bv);
          h1[(row0 + m_add + v) * 64 + nn] = (bf16_t)val;
        }
      }
    }
  }
  __syncthreads();

  // ---- GEMM2: [128x64] @ [64x32] -> 128x30 ----
  v8f b0 = {}, b1v = {};
  for (int ks = 0; ks < 2; ++ks) {
    v16bf af = load_a_frag(h1, row0, 64, ks * 32, lane);
    b0  = wmma_bf16(af, load_b_packed(pWb, ks * 2 + 0, lane), b0);
    b1v = wmma_bf16(af, load_b_packed(pWb, ks * 2 + 1, lane), b1v);
  }
  {
    v8f accs[2] = {b0, b1v};
#pragma unroll
    for (int t = 0; t < 2; ++t) {
      int nn = t * 16 + nlo;
      if (nn < 30) {
        float bv = bb[nn];
#pragma unroll
        for (int v = 0; v < 8; ++v) {
          float val = leaky(accs[t][v] + bv);
          h2[(row0 + m_add + v) * 32 + nn] = (bf16_t)val;
        }
      }
    }
  }
  __syncthreads();

  // ---- GEMM3: [128x32] @ [32x32] -> 128x20 (keep f32 + bf16 copies) ----
  {
    v16bf af = load_a_frag(h2, row0, 32, 0, lane);
    v8f z = {};
    v8f c0 = wmma_bf16(af, load_b_packed(pWc, 0, lane), z);
    v8f c1 = wmma_bf16(af, load_b_packed(pWc, 1, lane), z);
    v8f accs[2] = {c0, c1};
#pragma unroll
    for (int t = 0; t < 2; ++t) {
      int nn = t * 16 + nlo;
      if (nn < 20) {
        float bv = bc[nn];
#pragma unroll
        for (int v = 0; v < 8; ++v) {
          float val = leaky(accs[t][v] + bv);
          int r = row0 + m_add + v;
          h3f[r * 20 + nn] = val;
          h3b[r * 32 + nn] = (bf16_t)val;
        }
      }
    }
  }
  __syncthreads();

  // ---- GEMM4 (scores): [128x32] @ [32x32] -> 128x30, leaky ----
  {
    v16bf af = load_a_frag(h3b, row0, 32, 0, lane);
    v8f z = {};
    v8f d0 = wmma_bf16(af, load_b_packed(pP1, 0, lane), z);
    v8f d1 = wmma_bf16(af, load_b_packed(pP1, 1, lane), z);
    v8f accs[2] = {d0, d1};
#pragma unroll
    for (int t = 0; t < 2; ++t) {
      int nn = t * 16 + nlo;
      if (nn < 30) {
        float bv = p1b1[nn];
#pragma unroll
        for (int v = 0; v < 8; ++v)
          s1[(row0 + m_add + v) * 32 + nn] = leaky(accs[t][v] + bv);
      }
    }
  }
  __syncthreads();

  // Per-row score, softmax over T2=128, weighted pool -> 20 floats.
  if (tid < T2) {
    float s = p1b2[0];
#pragma unroll
    for (int j = 0; j < 30; ++j) s += s1[tid * 32 + j] * p1W2[j];
    sc[tid] = s;
  }
  __syncthreads();
  if (tid == 0) {
    float mx = sc[0];
    for (int t = 1; t < T2; ++t) mx = fmaxf(mx, sc[t]);
    float sum = 0.f;
    for (int t = 0; t < T2; ++t) {
      float e = __expf(sc[t] - mx);
      sc[t] = e;
      sum += e;
    }
    red[0] = sum;
  }
  __syncthreads();
  if (tid < 20) {
    float inv = 1.f / red[0];
    float acc = 0.f;
    for (int t = 0; t < T2; ++t) acc += sc[t] * h3f[t * 20 + tid];
    hpool[(size_t)blk * 20 + tid] = acc * inv;
  }
}

// ---------------- Stage B: pool over T1, normalize, head ----------------
__global__ __launch_bounds__(64) void stage_b_kernel(
    const float* __restrict__ hpool,
    const float* __restrict__ W1, const float* __restrict__ b1,
    const float* __restrict__ W2, const float* __restrict__ b2,
    const float* __restrict__ Wd, const float* __restrict__ bd,
    const float* __restrict__ We, const float* __restrict__ be,
    float* __restrict__ out) {
  __shared__ float rows[64 * 20];
  __shared__ float sc[64];
  __shared__ float red[2];
  __shared__ float h[20];
  const int tid = threadIdx.x;
  const int b   = blockIdx.x;

  for (int i = tid; i < 64 * 20; i += 64)
    rows[i] = hpool[(size_t)b * 64 * 20 + i];
  __syncthreads();

  {
    float s = b2[0];
    for (int j = 0; j < 30; ++j) {
      float u = b1[j];
      for (int c = 0; c < 20; ++c) u += rows[tid * 20 + c] * W1[c * 30 + j];
      u = leaky(u);
      s += u * W2[j];
    }
    sc[tid] = s;
  }
  __syncthreads();
  if (tid == 0) {
    float mx = sc[0];
    for (int t = 1; t < 64; ++t) mx = fmaxf(mx, sc[t]);
    float sum = 0.f;
    for (int t = 0; t < 64; ++t) {
      float e = __expf(sc[t] - mx);
      sc[t] = e;
      sum += e;
    }
    red[0] = sum;
  }
  __syncthreads();
  if (tid < 20) {
    float inv = 1.f / red[0];
    float a = 0.f;
    for (int t = 0; t < 64; ++t) a += sc[t] * rows[t * 20 + tid];
    h[tid] = a * inv;
  }
  __syncthreads();
  if (tid == 0) {
    float nrm = 0.f;
    for (int c = 0; c < 20; ++c) nrm += h[c] * h[c];
    nrm = fmaxf(sqrtf(nrm), 1e-12f);
    float o = be[0];
    for (int j = 0; j < 15; ++j) {
      float u = bd[j];
      for (int c = 0; c < 20; ++c) u += (h[c] / nrm) * Wd[c * 15 + j];
      u = leaky(u);
      o += u * We[j];
    }
    out[b] = o;
  }
}

extern "C" void kernel_launch(void* const* d_in, const int* in_sizes, int n_in,
                              void* d_out, int out_size, void* d_ws,
                              size_t ws_size, hipStream_t stream) {
  const float* x    = (const float*)d_in[0];
  const float* Wa   = (const float*)d_in[1];
  const float* ba   = (const float*)d_in[2];
  const float* Wb   = (const float*)d_in[3];
  const float* bb   = (const float*)d_in[4];
  const float* Wc   = (const float*)d_in[5];
  const float* bc   = (const float*)d_in[6];
  const float* p1W1 = (const float*)d_in[7];
  const float* p1b1 = (const float*)d_in[8];
  const float* p1W2 = (const float*)d_in[9];
  const float* p1b2 = (const float*)d_in[10];
  const float* p2W1 = (const float*)d_in[11];
  const float* p2b1 = (const float*)d_in[12];
  const float* p2W2 = (const float*)d_in[13];
  const float* p2b2 = (const float*)d_in[14];
  const float* Wd   = (const float*)d_in[15];
  const float* bd   = (const float*)d_in[16];
  const float* We   = (const float*)d_in[17];
  const float* be   = (const float*)d_in[18];

  float*  hpool = (float*)d_ws;                       // 4096 x 20 f32
  bf16_t* wpack = (bf16_t*)((char*)d_ws + 4096 * 20 * sizeof(float));
  float*  out   = (float*)d_out;                      // 64 f32

  pack_weights_kernel<<<1, NT, 0, stream>>>(Wa, Wb, Wc, p1W1, wpack);
  stage_a_kernel<<<64 * 64, NT, 0, stream>>>(x, wpack, ba, bb, bc, p1b1, p1W2,
                                             p1b2, hpool);
  stage_b_kernel<<<64, 64, 0, stream>>>(hpool, p2W1, p2b1, p2W2, p2b2, Wd, bd,
                                        We, be, out);
}
